// PatchAdapter_74483322848043
// MI455X (gfx1250) — matrix operand order repaired and checked by reading, not compiled, not run on hardware
//
#include <hip/hip_runtime.h>
#include <hip/hip_bf16.h>
#include <math.h>

// ---------------- problem constants ----------------
#define BATCH   4
#define CHN     64
#define IMG     256
#define PATCH   8
#define STR     4
#define NHW     63                    // (256-8)/4+1 ; reflect-pad rows are never read
#define LP      (NHW*NHW)             // 3969 patches per image
#define BL      (BATCH*LP)            // 15876 total patch rows
#define IN_DIM  4096                  // C*P*P
#define HID     256
#define MPAD    16000                 // 125 * 128 (GEMM M padding)

typedef __attribute__((ext_vector_type(16))) __bf16       v16bf;
typedef __attribute__((ext_vector_type(8)))  float        v8f;
typedef __attribute__((ext_vector_type(4)))  unsigned int u32x4;
typedef __attribute__((ext_vector_type(8)))  unsigned int u32x8;

// ---------------- weight conversion to bf16 ----------------
__global__ void k_cvt_weights(const float* __restrict__ W1, const float* __restrict__ W2,
                              __bf16* __restrict__ W1b, __bf16* __restrict__ W2b) {
    int i = blockIdx.x * blockDim.x + threadIdx.x;   // grid covers HID*IN_DIM = 1048576
    W1b[i] = (__bf16)W1[i];
    W2b[i] = (__bf16)W2[i];
}

// ---------------- im2col: x -> A (bf16, row = patch, col = c*64+kh*8+kw) ----------------
__global__ void k_im2col(const float* __restrict__ x, __bf16* __restrict__ Ab) {
    int m  = blockIdx.x;                 // patch row index in [0, BL)
    int b  = m / LP;
    int r  = m - b * LP;
    int py = r / NHW;
    int px = r - py * NHW;
    const float* xb   = x  + (size_t)b * CHN * IMG * IMG;
    __bf16*      arow = Ab + (size_t)m * IN_DIM;
    #pragma unroll
    for (int j = 0; j < IN_DIM / 256; ++j) {
        int k  = threadIdx.x + j * 256;
        int c  = k >> 6;
        int kh = (k >> 3) & 7;
        int kw = k & 7;
        float v = xb[((size_t)c * IMG + (py * STR + kh)) * IMG + (px * STR + kw)];
        arow[k] = (__bf16)v;
    }
}

// ---------------- CDNA5 async global->LDS copy (ASYNCcnt path) ----------------
// GVS mode: saddr = uniform 64-bit base (SGPR pair), vaddr = per-lane 32-bit byte
// offset, vdst = LDS byte address (low 32 bits of the generic shared pointer).
__device__ __forceinline__ void async_b128(unsigned lds_addr, unsigned voff, const void* sbase) {
    asm volatile("global_load_async_to_lds_b128 %0, %1, %2"
                 :: "v"(lds_addr), "v"(voff), "s"((unsigned long long)(size_t)sbase)
                 : "memory");
}
__device__ __forceinline__ void wait_async_6() { asm volatile("s_wait_asynccnt 0x6" ::: "memory"); }
__device__ __forceinline__ void wait_async_0() { asm volatile("s_wait_asynccnt 0x0" ::: "memory"); }

// Stage one [rows x 32] bf16 slab (4 16B chunks per row) into LDS asynchronously.
// ROW_STRIDE = global row stride in elements; CHUNKS = chunks per thread (rows*4/256).
template<int ROW_STRIDE, int CHUNKS>
__device__ __forceinline__ void issue_slab(const __bf16* gbase, unsigned kbyte,
                                           __bf16* lbase, int t) {
    #pragma unroll
    for (int i = 0; i < CHUNKS; ++i) {
        int ch  = t + i * 256;
        int row = ch >> 2, k8 = (ch & 3) * 8;
        unsigned l    = (unsigned)(size_t)(lbase + row * 32 + k8);
        unsigned voff = (unsigned)(row * ROW_STRIDE + k8) * 2u + kbyte;
        async_b128(l, voff, gbase);
    }
}

// ---------------- WMMA fragment loaders (LDS, contiguous k order, same for A and B) ----------------
// A 16x32 bf16: lanes 0-15 rows M=0..15; per lane two 16B chunks at k0=(lane>>4)*16 and k0+8
__device__ __forceinline__ v16bf load_a_frag(const __bf16* As, int row_base, int lane) {
    int m  = lane & 15;
    int k0 = (lane >> 4) * 16;
    const __bf16* p = As + (row_base + m) * 32;
    u32x4 lo = *reinterpret_cast<const u32x4*>(p + k0);
    u32x4 hi = *reinterpret_cast<const u32x4*>(p + k0 + 8);
    u32x8 u;
    u[0]=lo[0]; u[1]=lo[1]; u[2]=lo[2]; u[3]=lo[3];
    u[4]=hi[0]; u[5]=hi[1]; u[6]=hi[2]; u[7]=hi[3];
    return __builtin_bit_cast(v16bf, u);
}
// B 32x16 bf16: lane -> column n, lanes 0-15 K=0..15, lanes 16-31 K=16..31 (contiguous 32B)
__device__ __forceinline__ v16bf load_b_frag(const __bf16* Ws, int col_base, int lane) {
    int n  = col_base + (lane & 15);
    int kb = (lane >> 4) * 16;
    u32x8 u = *reinterpret_cast<const u32x8*>(Ws + n * 32 + kb);
    return __builtin_bit_cast(v16bf, u);
}

__device__ __forceinline__ float gelu_exact(float v) {
    return 0.5f * v * (1.0f + erff(v * 0.70710678118654752f));
}

// ---------------- GEMM1: Hm = gelu(A @ W1^T + b1), bf16 out ----------------
// block = 256 thr (8 waves as 2x4), tile 128(M) x 256(N=HID), K slab = 32, double-buffered LDS
__global__ void __launch_bounds__(256) k_gemm1(const __bf16* __restrict__ Ab,
                                               const __bf16* __restrict__ W1b,
                                               const float*  __restrict__ b1,
                                               __bf16* __restrict__ Hb) {
    __shared__ alignas(16) __bf16 As[2][128 * 32];
    __shared__ alignas(16) __bf16 Ws[2][256 * 32];
    int t = threadIdx.x, wave = t >> 5, lane = t & 31;
    int wm = wave >> 2, wn = wave & 3;
    int mBase = blockIdx.x * 128;
    const __bf16* Ag = Ab + (size_t)mBase * IN_DIM;
    constexpr int NK = IN_DIM / 32;      // 128

    v8f zero = {0.f,0.f,0.f,0.f,0.f,0.f,0.f,0.f};
    v8f acc[4][4];
    #pragma unroll
    for (int i = 0; i < 4; ++i)
        #pragma unroll
        for (int j = 0; j < 4; ++j) acc[i][j] = zero;

    // prologue: slabs 0 and 1 in flight (12 async instrs/wave)
    issue_slab<IN_DIM, 2>(Ag,  0,  As[0], t);
    issue_slab<IN_DIM, 4>(W1b, 0,  Ws[0], t);
    issue_slab<IN_DIM, 2>(Ag,  64, As[1], t);
    issue_slab<IN_DIM, 4>(W1b, 64, Ws[1], t);
    wait_async_6();                      // slab 0 resident (async loads retire in order)
    __syncthreads();

    for (int kt = 0; kt < NK; ++kt) {
        int buf = kt & 1;
        v16bf afrag[4];
        #pragma unroll
        for (int mi = 0; mi < 4; ++mi) afrag[mi] = load_a_frag(As[buf], wm * 64 + mi * 16, lane);
        #pragma unroll
        for (int ni = 0; ni < 4; ++ni) {
            v16bf bfrag = load_b_frag(Ws[buf], wn * 64 + ni * 16, lane);
            #pragma unroll
            for (int mi = 0; mi < 4; ++mi)
                acc[mi][ni] = __builtin_amdgcn_wmma_f32_16x16x32_bf16(
                    false, afrag[mi], false, bfrag, (short)0, acc[mi][ni], false, false);
        }
        if (kt + 1 < NK) {
            __syncthreads();             // everyone done reading buf -> safe to refill it
            if (kt + 2 < NK) {
                unsigned kb = (unsigned)(kt + 2) * 64u;
                issue_slab<IN_DIM, 2>(Ag,  kb, As[buf], t);
                issue_slab<IN_DIM, 4>(W1b, kb, Ws[buf], t);
                wait_async_6();          // slab kt+1 resident
            } else {
                wait_async_0();
            }
            __syncthreads();
        }
    }

    int half = lane >> 4, nlane = lane & 15;
    #pragma unroll
    for (int mi = 0; mi < 4; ++mi)
        #pragma unroll
        for (int ni = 0; ni < 4; ++ni) {
            int col = wn * 64 + ni * 16 + nlane;
            float bb = b1[col];
            #pragma unroll
            for (int i = 0; i < 8; ++i) {
                int rl = wm * 64 + mi * 16 + i + half * 8;
                float g = gelu_exact(acc[mi][ni][i] + bb);
                Hb[(size_t)(mBase + rl) * HID + col] = (__bf16)g;
            }
        }
}

// ---------------- GEMM2 + fused fold scatter: acc += Hm @ W2^T + b2 ----------------
// grid (125, 16): tile 128(M) x 256(N of 4096), K = HID = 256 in 8 slabs of 32
__global__ void __launch_bounds__(256) k_gemm2(const __bf16* __restrict__ Hb,
                                               const __bf16* __restrict__ W2b,
                                               const float*  __restrict__ b2,
                                               float* __restrict__ foldacc) {
    __shared__ alignas(16) __bf16 As[2][128 * 32];
    __shared__ alignas(16) __bf16 Ws[2][256 * 32];
    int t = threadIdx.x, wave = t >> 5, lane = t & 31;
    int wm = wave >> 2, wn = wave & 3;
    int mBase = blockIdx.x * 128;
    int nBase = blockIdx.y * 256;
    const __bf16* Ag = Hb  + (size_t)mBase * HID;
    const __bf16* Wg = W2b + (size_t)nBase * HID;
    constexpr int NK = HID / 32;         // 8

    v8f zero = {0.f,0.f,0.f,0.f,0.f,0.f,0.f,0.f};
    v8f acc[4][4];
    #pragma unroll
    for (int i = 0; i < 4; ++i)
        #pragma unroll
        for (int j = 0; j < 4; ++j) acc[i][j] = zero;

    issue_slab<HID, 2>(Ag, 0,  As[0], t);
    issue_slab<HID, 4>(Wg, 0,  Ws[0], t);
    issue_slab<HID, 2>(Ag, 64, As[1], t);
    issue_slab<HID, 4>(Wg, 64, Ws[1], t);
    wait_async_6();
    __syncthreads();

    for (int kt = 0; kt < NK; ++kt) {
        int buf = kt & 1;
        v16bf afrag[4];
        #pragma unroll
        for (int mi = 0; mi < 4; ++mi) afrag[mi] = load_a_frag(As[buf], wm * 64 + mi * 16, lane);
        #pragma unroll
        for (int ni = 0; ni < 4; ++ni) {
            v16bf bfrag = load_b_frag(Ws[buf], wn * 64 + ni * 16, lane);
            #pragma unroll
            for (int mi = 0; mi < 4; ++mi)
                acc[mi][ni] = __builtin_amdgcn_wmma_f32_16x16x32_bf16(
                    false, afrag[mi], false, bfrag, (short)0, acc[mi][ni], false, false);
        }
        if (kt + 1 < NK) {
            __syncthreads();
            if (kt + 2 < NK) {
                unsigned kb = (unsigned)(kt + 2) * 64u;
                issue_slab<HID, 2>(Ag, kb, As[buf], t);
                issue_slab<HID, 4>(Wg, kb, Ws[buf], t);
                wait_async_6();
            } else {
                wait_async_0();
            }
            __syncthreads();
        }
    }

    // epilogue: y = acc + b2[col]; scatter-add into fold accumulator (<=4 hits per pixel)
    int half = lane >> 4, nlane = lane & 15;
    #pragma unroll
    for (int mi = 0; mi < 4; ++mi) {
        #pragma unroll
        for (int i = 0; i < 8; ++i) {
            int m = mBase + wm * 64 + mi * 16 + i + half * 8;
            if (m < BL) {
                int b  = m / LP;
                int r  = m - b * LP;
                int py = r / NHW;
                int px = r - py * NHW;
                #pragma unroll
                for (int ni = 0; ni < 4; ++ni) {
                    int col = nBase + wn * 64 + ni * 16 + nlane;
                    int c  = col >> 6;
                    int kh = (col >> 3) & 7;
                    int kw = col & 7;
                    float y = acc[mi][ni][i] + b2[col];
                    int h = py * STR + kh, w = px * STR + kw;
                    atomicAdd(foldacc + (((size_t)b * CHN + c) * IMG + h) * IMG + w, y);
                }
            }
        }
    }
}

// ---------------- finalize: out = x + softplus(alpha) * acc / cnt ----------------
// fold(unfold(x))/norm == x exactly, so only fold(y) needs normalization.
__global__ void k_final(const float* __restrict__ x, const float* __restrict__ foldacc,
                        const float* __restrict__ alpha_raw, float* __restrict__ out, int n) {
    int i = blockIdx.x * blockDim.x + threadIdx.x;
    if (i >= n) return;
    float alpha = log1pf(expf(alpha_raw[0]));   // softplus
    int w = i & 255, h = (i >> 8) & 255;
    float cH = (h < 4 || h >= 252) ? 1.0f : 2.0f;
    float cW = (w < 4 || w >= 252) ? 1.0f : 2.0f;
    out[i] = x[i] + alpha * foldacc[i] / (cH * cW);
}

// ---------------- launcher ----------------
extern "C" void kernel_launch(void* const* d_in, const int* in_sizes, int n_in,
                              void* d_out, int out_size, void* d_ws, size_t ws_size,
                              hipStream_t stream) {
    const float* x         = (const float*)d_in[0];
    const float* W1        = (const float*)d_in[1];
    const float* b1        = (const float*)d_in[2];
    const float* W2        = (const float*)d_in[3];
    const float* b2        = (const float*)d_in[4];
    const float* alpha_raw = (const float*)d_in[5];
    float* out = (float*)d_out;

    char* ws = (char*)d_ws;
    const size_t SZ_W   = (size_t)HID * IN_DIM * sizeof(__bf16);   // 2 MB each
    const size_t SZ_A   = (size_t)MPAD * IN_DIM * sizeof(__bf16);  // 131 MB
    const size_t SZ_H   = (size_t)MPAD * HID   * sizeof(__bf16);   // 8 MB
    const size_t SZ_ACC = (size_t)BATCH * CHN * IMG * IMG * sizeof(float); // 67 MB
    __bf16* W1b = (__bf16*)(ws);
    __bf16* W2b = (__bf16*)(ws + SZ_W);
    __bf16* Ab  = (__bf16*)(ws + 2 * SZ_W);
    __bf16* Hb  = (__bf16*)(ws + 2 * SZ_W + SZ_A);
    float*  acc = (float*) (ws + 2 * SZ_W + SZ_A + SZ_H);

    hipMemsetAsync(acc, 0, SZ_ACC, stream);

    k_cvt_weights<<<(HID * IN_DIM) / 256, 256, 0, stream>>>(W1, W2, W1b, W2b);
    k_im2col<<<BL, 256, 0, stream>>>(x, Ab);
    k_gemm1<<<MPAD / 128, 256, 0, stream>>>(Ab, W1b, b1, Hb);
    dim3 g2(MPAD / 128, IN_DIM / 256);
    k_gemm2<<<g2, 256, 0, stream>>>(Hb, W2b, b2, acc);
    int n = BATCH * CHN * IMG * IMG;
    k_final<<<(n + 255) / 256, 256, 0, stream>>>(x, acc, alpha_raw, out, n);
}
